// IDCST2_1133871366835
// MI455X (gfx1250) — matrix operand order repaired and checked by reading, not compiled, not run on hardware
//
#include <hip/hip_runtime.h>

// IDCST2: out = C0 @ (x @ S1^T), all 4096x4096 fp32.
// bf16x3 split-precision GEMMs on v_wmma_f32_16x16x32_bf16, double-buffered LDS
// with CDNA5 async global->LDS copies (ASYNCcnt).
// Workspace layout (needs 8 * 32MB = 256MB):
//   Xhi Xlo | SBhi SBlo (S1 as Bt [v][q]) | CAhi CAlo (C0 [u][p]) | Yhi Ylo (y as Bt [v][p])

#define MD 4096
#define ND 4096
#define KD 4096

typedef __attribute__((ext_vector_type(16))) __bf16        v16bf;
typedef __attribute__((ext_vector_type(8)))  float         v8f;
typedef __attribute__((ext_vector_type(4)))  unsigned int  v4u;
typedef __attribute__((ext_vector_type(4)))  int           v4i;

union Frag { v16bf v; v4u q[2]; };

#if __has_builtin(__builtin_amdgcn_global_load_async_to_lds_b128)
#define ASYNC_OK 1
#else
#define ASYNC_OK 0
#endif

#if ASYNC_OK
__device__ __forceinline__ void async_cp16(const void* g, void* l) {
  __builtin_amdgcn_global_load_async_to_lds_b128(
      (__attribute__((address_space(1))) v4i*)g,
      (__attribute__((address_space(3))) v4i*)l, 0, 0);
}
#if __has_builtin(__builtin_amdgcn_s_wait_asynccnt)
#define WAIT_TILE(n) __builtin_amdgcn_s_wait_asynccnt(n)
#else
#define WAIT_TILE(n) asm volatile("s_wait_asynccnt %0" ::"i"(n) : "memory")
#endif
#else
#define WAIT_TILE(n)
#endif

__device__ __forceinline__ unsigned short f2bf(float f) {
  unsigned u = __float_as_uint(f);
  u += 0x7FFFu + ((u >> 16) & 1u);   // round-to-nearest-even
  return (unsigned short)(u >> 16);
}
__device__ __forceinline__ float bf2f(unsigned short b) {
  return __uint_as_float((unsigned)b << 16);
}

// ---------------- prep kernels ----------------

__global__ void split_f32_kernel(const float* __restrict__ x,
                                 unsigned short* __restrict__ hi,
                                 unsigned short* __restrict__ lo) {
  size_t i = (size_t)blockIdx.y * 4096 + (size_t)blockIdx.x * 256 + threadIdx.x;
  float f = x[i];
  unsigned short h = f2bf(f);
  hi[i] = h;
  lo[i] = f2bf(f - bf2f(h));
}

// SBt[v][q] = sin(pi*(2v+1)*q / (2N)); exact integer phase reduction mod 4N.
__global__ void gen_sin_bt_kernel(unsigned short* __restrict__ hi,
                                  unsigned short* __restrict__ lo) {
  int q = blockIdx.x * 256 + threadIdx.x;
  int v = blockIdx.y;
  int t = ((2 * v + 1) * q) & (4 * ND - 1);     // mod 16384
  float s = sinf((float)t * 3.8349519697e-4f);  // pi/8192
  unsigned short h = f2bf(s);
  size_t i = (size_t)v * 4096 + q;
  hi[i] = h;
  lo[i] = f2bf(s - bf2f(h));
}

// CA[u][p] = cos(pi*(2u+1)*p / (2M))
__global__ void gen_cos_a_kernel(unsigned short* __restrict__ hi,
                                 unsigned short* __restrict__ lo) {
  int p = blockIdx.x * 256 + threadIdx.x;
  int u = blockIdx.y;
  int t = ((2 * u + 1) * p) & (4 * MD - 1);
  float c = cosf((float)t * 3.8349519697e-4f);
  unsigned short h = f2bf(c);
  size_t i = (size_t)u * 4096 + p;
  hi[i] = h;
  lo[i] = f2bf(c - bf2f(h));
}

// ---------------- bf16x3 WMMA GEMM ----------------
// A: [M][K] row-major bf16 hi/lo.  Bt: [N][K] row-major bf16 hi/lo.
// Block tile 128x128, K-step 32, double-buffered. 256 threads = 8 waves (wave32),
// wave grid 2(M) x 4(N), each wave 4x2 tiles of 16x16 -> 24 v_wmma per K-step.
// EPI==0: fp32 out[m][n] (stride ND). EPI==1: bf16 hi/lo transposed out[n][m].

template <int EPI>
__global__ __launch_bounds__(256)
void gemm_bf16x3_kernel(const unsigned short* __restrict__ Ahi,
                        const unsigned short* __restrict__ Alo,
                        const unsigned short* __restrict__ Bthi,
                        const unsigned short* __restrict__ Btlo,
                        float* __restrict__ outF,
                        unsigned short* __restrict__ outHi,
                        unsigned short* __restrict__ outLo) {
  // Row = 32 bf16 = 64B = 4 x b128.  [buf][hi/lo][row*4 + chunk]
  __shared__ v4u sA[2][2][128 * 4];
  __shared__ v4u sB[2][2][128 * 4];

  const int t     = threadIdx.x;
  const int bn    = blockIdx.x;
  const int bm    = blockIdx.y;
  const int wave  = t >> 5;       // wave32
  const int lane  = t & 31;
  const int wm    = wave >> 2;    // 0..1  (64 rows each)
  const int wn    = wave & 3;     // 0..3  (32 cols each)
  const int lhalf = lane >> 4;    // lanes 16..31 -> second half per ISA layout
  const int lmod  = lane & 15;
  const int chunk = t & 3;        // 16B chunk within a 64B row
  const int srow  = t >> 2;       // 0..63

  // Stage one 128x32 K-tile of A and Bt (hi+lo): 8 x b128 per thread.
  auto stage = [&](int buf, int ktile) {
    const int kof = ktile * 32;
    for (int it = 0; it < 2; ++it) {
      int r = srow + it * 64;
      int idx = r * 4 + chunk;
      size_t ga = (size_t)(bm * 128 + r) * KD + kof + chunk * 8;
      size_t gb = (size_t)(bn * 128 + r) * KD + kof + chunk * 8;
#if ASYNC_OK
      async_cp16(Ahi + ga, &sA[buf][0][idx]);   // global_load_async_to_lds_b128
      async_cp16(Alo + ga, &sA[buf][1][idx]);
      async_cp16(Bthi + gb, &sB[buf][0][idx]);
      async_cp16(Btlo + gb, &sB[buf][1][idx]);
#else
      sA[buf][0][idx] = *(const v4u*)(Ahi + ga);
      sA[buf][1][idx] = *(const v4u*)(Alo + ga);
      sB[buf][0][idx] = *(const v4u*)(Bthi + gb);
      sB[buf][1][idx] = *(const v4u*)(Btlo + gb);
      if (ktile + 1 < KD / 32) {  // global_prefetch_b8 of next K tile
        __builtin_prefetch(Ahi + ga + 32, 0, 1);
        __builtin_prefetch(Bthi + gb + 32, 0, 1);
      }
#endif
    }
  };

  v8f zero = {};
  v8f acc[4][2];
  for (int mt = 0; mt < 4; ++mt)
    for (int nt = 0; nt < 2; ++nt) acc[mt][nt] = zero;

  stage(0, 0);

  for (int kt = 0; kt < KD / 32; ++kt) {
    const int cur = kt & 1;
    if (kt + 1 < KD / 32) {
      stage(cur ^ 1, kt + 1);   // prefetch next tile into other buffer (async)
      WAIT_TILE(8);             // in-order: current tile's 8 loads complete
    } else {
      WAIT_TILE(0);
    }
    __syncthreads();            // all waves' current-tile LDS writes visible

    const v4u* A0 = sA[cur][0];
    const v4u* A1 = sA[cur][1];
    const v4u* B0 = sB[cur][0];
    const v4u* B1 = sB[cur][1];

    // A fragments: 16x32 bf16; lane m = lmod, VGPR j<4: K=8*lhalf+2j, j>=4: +16
    Frag ah[4], al[4];
    for (int mt = 0; mt < 4; ++mt) {
      int m4 = (wm * 64 + mt * 16 + lmod) * 4;
      ah[mt].q[0] = A0[m4 + lhalf];
      ah[mt].q[1] = A0[m4 + 2 + lhalf];
      al[mt].q[0] = A1[m4 + lhalf];
      al[mt].q[1] = A1[m4 + 2 + lhalf];
    }

    // B fragments: 32x16 bf16; lane n = lmod, VGPR j<4: K=8*lhalf+2j, j>=4: +16
    for (int nt = 0; nt < 2; ++nt) {
      int n4 = (wn * 32 + nt * 16 + lmod) * 4;
      Frag bh, bl;
      bh.q[0] = B0[n4 + lhalf];
      bh.q[1] = B0[n4 + 2 + lhalf];
      bl.q[0] = B1[n4 + lhalf];
      bl.q[1] = B1[n4 + 2 + lhalf];

      for (int mt = 0; mt < 4; ++mt) {
        acc[mt][nt] = __builtin_amdgcn_wmma_f32_16x16x32_bf16(
            false, ah[mt].v, false, bh.v, (short)0, acc[mt][nt], false, false);
        acc[mt][nt] = __builtin_amdgcn_wmma_f32_16x16x32_bf16(
            false, ah[mt].v, false, bl.v, (short)0, acc[mt][nt], false, false);
        acc[mt][nt] = __builtin_amdgcn_wmma_f32_16x16x32_bf16(
            false, al[mt].v, false, bh.v, (short)0, acc[mt][nt], false, false);
      }
    }
    __syncthreads();            // reads of `cur` done; next iter may overwrite it
  }

  // epilogue: C/D layout: lane n = lmod, VGPR j -> m = j + 8*lhalf
  for (int mt = 0; mt < 4; ++mt) {
    for (int nt = 0; nt < 2; ++nt) {
      int mbase = bm * 128 + wm * 64 + mt * 16 + 8 * lhalf;
      int n     = bn * 128 + wn * 32 + nt * 16 + lmod;
      if (EPI == 0) {
        for (int j = 0; j < 8; ++j)
          outF[(size_t)(mbase + j) * ND + n] = acc[mt][nt][j];
      } else {
        union { v4u q; unsigned short s[8]; } ph, pl;
        for (int j = 0; j < 8; ++j) {
          float f = acc[mt][nt][j];
          unsigned short h = f2bf(f);
          ph.s[j] = h;
          pl.s[j] = f2bf(f - bf2f(h));
        }
        *((v4u*)(outHi + (size_t)n * MD + mbase)) = ph.q;  // transposed [n][m]
        *((v4u*)(outLo + (size_t)n * MD + mbase)) = pl.q;
      }
    }
  }
}

// ---------------- launch ----------------

extern "C" void kernel_launch(void* const* d_in, const int* in_sizes, int n_in,
                              void* d_out, int out_size, void* d_ws, size_t ws_size,
                              hipStream_t stream) {
  (void)in_sizes; (void)n_in; (void)out_size; (void)ws_size;
  const float* x = (const float*)d_in[0];
  float* out = (float*)d_out;

  const size_t S = (size_t)4096 * 4096;  // elements per matrix
  unsigned short* p    = (unsigned short*)d_ws;
  unsigned short* Xhi  = p + 0 * S;
  unsigned short* Xlo  = p + 1 * S;
  unsigned short* SBhi = p + 2 * S;
  unsigned short* SBlo = p + 3 * S;
  unsigned short* CAhi = p + 4 * S;
  unsigned short* CAlo = p + 5 * S;
  unsigned short* Yhi  = p + 6 * S;
  unsigned short* Ylo  = p + 7 * S;

  dim3 pb(256), pg(16, 4096);
  split_f32_kernel<<<pg, pb, 0, stream>>>(x, Xhi, Xlo);
  gen_sin_bt_kernel<<<pg, pb, 0, stream>>>(SBhi, SBlo);
  gen_cos_a_kernel<<<pg, pb, 0, stream>>>(CAhi, CAlo);

  dim3 gg(32, 32), gb(256);
  // GEMM1: y = x @ S1^T, written transposed as bf16 hi/lo (B operand of GEMM2)
  gemm_bf16x3_kernel<1><<<gg, gb, 0, stream>>>(Xhi, Xlo, SBhi, SBlo,
                                               nullptr, Yhi, Ylo);
  // GEMM2: out = C0 @ y, fp32 output
  gemm_bf16x3_kernel<0><<<gg, gb, 0, stream>>>(CAhi, CAlo, Yhi, Ylo,
                                               out, nullptr, nullptr);
}
